// RefineModel_42056319762453
// MI455X (gfx1250) — matrix operand order repaired and checked by reading, not compiled, not run on hardware
//
#include <hip/hip_runtime.h>
#include <hip/hip_bf16.h>
#include <math.h>

typedef __bf16 bf16_t;
typedef __attribute__((ext_vector_type(16))) __bf16 v16bf;
typedef __attribute__((ext_vector_type(8)))  float  v8f;

#define K_   16
#define T_   40
#define B_   16
#define N_   32
#define BN_  512
#define KBN_ 8192

#define PI_F 3.14159265358979323846f
#define R0_F 0.5f
#define R1_F 4.0f
#define R_STEP  ((R1_F - R0_F) / 6.0f)
#define TH_STEP (2.0f * PI_F / 6.0f)

// ---------------------------------------------------------------------------
// Generic WMMA GEMM:  C[z] = A (MxKpad) * Bt[z]^T (NxKpad) + bias
// One wave (32 threads) per 16x16 output tile. bf16 in, f32 accumulate.
// Fragment layouts per CDNA5 ISA 7.12.2 (wave32).
// ---------------------------------------------------------------------------
__global__ void wmma_gemm_bf16(const bf16_t* __restrict__ A, int lda,
                               const bf16_t* __restrict__ Bt, int ldb, long bz,
                               const float* __restrict__ bias,
                               float* __restrict__ C, int ldc, long cz,
                               int Kpad)
{
    const int lane = threadIdx.x;          // 0..31
    const int m0 = blockIdx.x * 16;
    const int n0 = blockIdx.y * 16;
    Bt += (long)blockIdx.z * bz;
    C  += (long)blockIdx.z * cz;

    const int mrow = lane & 15;            // row within tile (A) / col (B,C)
    const int hi   = lane >> 4;            // lane half-group

    v8f acc = {};
    for (int k0 = 0; k0 < Kpad; k0 += 32) {
        v16bf a, b;
        // A fragment: lane l holds row (m0+mrow); K packed per ISA table.
        const bf16_t* Arow = A + (long)(m0 + mrow) * lda + k0;
#pragma unroll
        for (int g = 0; g < 2; ++g) {
#pragma unroll
            for (int p = 0; p < 4; ++p) {
                const int kk = g * 16 + hi * 8 + p * 2;
                a[g * 8 + p * 2]     = Arow[kk];
                a[g * 8 + p * 2 + 1] = Arow[kk + 1];
            }
        }
        // B fragment: B(k,n) = Bt[n][k]; K packed per ISA table (same as A).
        const bf16_t* Brow = Bt + (long)(n0 + mrow) * ldb + k0 + hi * 8;
#pragma unroll
        for (int v = 0; v < 8; ++v) {
            const int kb = (v < 4) ? 2 * v : 2 * v + 8;
            b[2 * v]     = Brow[kb];
            b[2 * v + 1] = Brow[kb + 1];
        }
        acc = __builtin_amdgcn_wmma_f32_16x16x32_bf16(
            false, a, false, b, (short)0, acc, false, false);
    }
    // D layout: VGPR r -> row m0 + hi*8 + r, col n0 + mrow
    const int n  = n0 + mrow;
    const int mb = m0 + hi * 8;
    const float bv = bias ? bias[n] : 0.0f;
#pragma unroll
    for (int r = 0; r < 8; ++r)
        C[(long)(mb + r) * ldc + n] = acc[r] + bv;
}

// ---------------------------------------------------------------------------
// One-time weight staging: fp32 -> zero-padded bf16 (K padded to mult of 32)
// ---------------------------------------------------------------------------
__global__ void k_prep_w(const float* __restrict__ scf, const float* __restrict__ wih,
                         const float* __restrict__ whh, const float* __restrict__ wdy,
                         bf16_t* Wscf, bf16_t* Wih, bf16_t* Whh, bf16_t* Wdy)
{
    int i = blockIdx.x * blockDim.x + threadIdx.x;
    // Wscf[o][c][h(64)] = fc_scf_w[c][o*48+h]   (36*48*64)
    if (i < 36 * 48 * 64) {
        int h = i & 63, c = (i >> 6) % 48, o = i / (48 * 64);
        Wscf[i] = (h < 48) ? (bf16_t)scf[c * 1728 + o * 48 + h] : (bf16_t)0.0f;
        return;
    }
    i -= 36 * 48 * 64;
    if (i < 144 * 96) { Wih[i] = (bf16_t)wih[i]; return; }   // Kpad==K==96
    i -= 144 * 96;
    if (i < 144 * 64) {                                      // 48 -> 64 pad
        int k = i & 63, c = i >> 6;
        Whh[i] = (k < 48) ? (bf16_t)whh[c * 48 + k] : (bf16_t)0.0f;
        return;
    }
    i -= 144 * 64;
    if (i < 80 * 64) {                                       // 48 -> 64 pad
        int k = i & 63, c = i >> 6;
        Wdy[i] = (k < 48) ? (bf16_t)wdy[c * 48 + k] : (bf16_t)0.0f;
    }
}

// ---------------------------------------------------------------------------
// h init (tile hx K times) + zero score accumulator
// ---------------------------------------------------------------------------
__global__ void k_init(const float* __restrict__ hx, float* __restrict__ h,
                       float* __restrict__ score)
{
    int i = blockIdx.x * blockDim.x + threadIdx.x;
    if (i < KBN_ * 48) {
        int bn = (i / 48) & (BN_ - 1);
        h[i] = hx[bn * 48 + (i % 48)];
    }
    if (i < KBN_) score[i] = 0.0f;
}

// fp32 h (KBN x 48) -> bf16 zero-padded (KBN x 64). Rows 0..511 double as Hn.
__global__ void k_h2bf16(const float* __restrict__ h, bf16_t* __restrict__ hb)
{
    int i = blockIdx.x * blockDim.x + threadIdx.x;
    if (i >= KBN_ * 64) return;
    int row = i >> 6, c = i & 63;
    hb[i] = (c < 48) ? (bf16_t)h[row * 48 + c] : (bf16_t)0.0f;
}

// ---------------------------------------------------------------------------
// conv1: (16,4,160,160) -> relu -> (16,16,80,80), k=5 s=2 p=2
// ---------------------------------------------------------------------------
__global__ void k_conv1(const float* __restrict__ img, const float* __restrict__ w,
                        const float* __restrict__ bias, float* __restrict__ out)
{
    int i = blockIdx.x * blockDim.x + threadIdx.x;
    if (i >= B_ * 16 * 80 * 80) return;
    int ox = i % 80, oy = (i / 80) % 80, oc = (i / 6400) % 16, b = i / (16 * 6400);
    float acc = bias[oc];
    for (int ic = 0; ic < 4; ++ic) {
        const float* ip = img + (long)(b * 4 + ic) * 160 * 160;
        const float* wp = w + (oc * 4 + ic) * 25;
#pragma unroll
        for (int ky = 0; ky < 5; ++ky) {
            int iy = oy * 2 - 2 + ky;
            if (iy < 0 || iy >= 160) continue;
#pragma unroll
            for (int kx = 0; kx < 5; ++kx) {
                int ix = ox * 2 - 2 + kx;
                if (ix < 0 || ix >= 160) continue;
                acc += ip[iy * 160 + ix] * wp[ky * 5 + kx];
            }
        }
    }
    out[i] = fmaxf(acc, 0.0f);
}

// conv2: (16,16,80,80) -> relu -> fm NHWC (16,80,80,32), k=5 s=1 p=2
__global__ void k_conv2(const float* __restrict__ in, const float* __restrict__ w,
                        const float* __restrict__ bias, float* __restrict__ fm)
{
    int i = blockIdx.x * blockDim.x + threadIdx.x;
    if (i >= B_ * 32 * 80 * 80) return;
    int ox = i % 80, oy = (i / 80) % 80, oc = (i / 6400) % 32, b = i / (32 * 6400);
    float acc = bias[oc];
    for (int ic = 0; ic < 16; ++ic) {
        const float* ip = in + (long)(b * 16 + ic) * 80 * 80;
        const float* wp = w + (oc * 16 + ic) * 25;
#pragma unroll
        for (int ky = 0; ky < 5; ++ky) {
            int iy = oy - 2 + ky;
            if (iy < 0 || iy >= 80) continue;
#pragma unroll
            for (int kx = 0; kx < 5; ++kx) {
                int ix = ox - 2 + kx;
                if (ix < 0 || ix >= 80) continue;
                acc += ip[iy * 80 + ix] * wp[ky * 5 + kx];
            }
        }
    }
    fm[(((long)b * 80 + oy) * 80 + ox) * 32 + oc] = fmaxf(acc, 0.0f);
}

// ---------------------------------------------------------------------------
// Social-pooling bins: per (k,b,j) compute bin index + 1/cnt weight per t
// ---------------------------------------------------------------------------
__global__ void k_bins(const float* __restrict__ y_path, int tstep,
                       float* __restrict__ Aw, int* __restrict__ Bi)
{
    int idx = blockIdx.x * blockDim.x + threadIdx.x;
    if (idx >= KBN_) return;
    int j = idx & 31, b = (idx >> 5) & 15, kk = idx >> 9;
    const float* Pb = y_path + ((long)(kk * T_ + tstep) * BN_ + (long)b * N_) * 2;
    float xj = Pb[j * 2], yj = Pb[j * 2 + 1];
    int bins[32];
    int cnt[36];
#pragma unroll
    for (int o = 0; o < 36; ++o) cnt[o] = 0;
    for (int t = 0; t < 32; ++t) {
        float cx = Pb[t * 2] - xj, cy = Pb[t * 2 + 1] - yj;
        float dist = sqrtf(cx * cx + cy * cy);
        bool mask = (dist >= R0_F) && (dist <= R1_F) && (t != j);
        float dc = dist < 1e-10f ? 1e-10f : dist;
        float ct = fminf(1.0f, fmaxf(-1.0f, cx / dc));
        float costh = acosf(ct);
        float theta = (cy < -0.01f) ? (2.0f * PI_F - costh) : costh;
        int ub = (int)((dist - R0_F) / R_STEP); ub = ub < 0 ? 0 : (ub > 5 ? 5 : ub);
        int vb = (int)(theta / TH_STEP);        vb = vb < 0 ? 0 : (vb > 5 ? 5 : vb);
        int bb = ub * 6 + vb;
        bins[t] = mask ? bb : -1;
        if (mask) cnt[bb]++;
    }
    for (int t = 0; t < 32; ++t) {
        int bb = bins[t];
        Aw[(long)idx * 32 + t] = (bb >= 0) ? 1.0f / (float)cnt[bb] : 0.0f;
        Bi[(long)idx * 32 + t] = bb < 0 ? 0 : bb;
    }
}

// ---------------------------------------------------------------------------
// lhalf: map-feature gather (32) + velocity fc (16) -> x_bf16 cols 0..47
// ---------------------------------------------------------------------------
__global__ void k_lhalf(const float* __restrict__ y_path, const float* __restrict__ cur,
                        const float* __restrict__ fm, const float* __restrict__ fvw,
                        const float* __restrict__ fvb, bf16_t* __restrict__ x, int tstep)
{
    int idx = blockIdx.x * blockDim.x + threadIdx.x;
    if (idx >= KBN_) return;
    int bn = idx & (BN_ - 1), kk = idx >> 9, b = bn >> 5;
    const float* P = y_path + ((long)(kk * T_ + tstep) * BN_ + bn) * 2;
    float px = P[0], py = P[1];
    int u = 40 - (int)py; u = u < 0 ? 0 : (u > 79 ? 79 : u);
    int v = (int)px;       v = v < 0 ? 0 : (v > 79 ? 79 : v);
    const float* f = fm + (((long)b * 80 + u) * 80 + v) * 32;
    bf16_t* xr = x + (long)idx * 96;
#pragma unroll
    for (int c = 0; c < 32; ++c) xr[c] = (bf16_t)f[c];
    float pxp, pyp;
    if (tstep == 0) { pxp = cur[bn * 2]; pyp = cur[bn * 2 + 1]; }
    else {
        const float* Pp = y_path + ((long)(kk * T_ + tstep - 1) * BN_ + bn) * 2;
        pxp = Pp[0]; pyp = Pp[1];
    }
    float vx = (px - pxp) * 10.0f, vy = (py - pyp) * 10.0f;
#pragma unroll
    for (int c = 0; c < 16; ++c)
        xr[32 + c] = (bf16_t)(vx * fvw[c * 2] + vy * fvw[c * 2 + 1] + fvb[c]);
}

// rhalf: gather-weighted sum of precomputed G -> x_bf16 cols 48..95
__global__ void k_rhalf(const float* __restrict__ Aw, const int* __restrict__ Bi,
                        const float* __restrict__ G, const float* __restrict__ scf_b,
                        bf16_t* __restrict__ x)
{
    int row = blockIdx.x;      // k*512 + b*32 + j
    int c = threadIdx.x;       // 64 threads, first 48 active for output
    __shared__ float sA[32];
    __shared__ int   sB[32];
    if (c < 32) { sA[c] = Aw[(long)row * 32 + c]; sB[c] = Bi[(long)row * 32 + c]; }
    __syncthreads();
    if (c >= 48) return;
    int b = (row >> 5) & 15;
    float acc = scf_b[c];
    for (int t = 0; t < 32; ++t) {
        float w = sA[t];
        if (w != 0.0f)
            acc += w * G[((long)sB[t] * 512 + b * 32 + t) * 48 + c];
    }
    x[(long)row * 96 + 48 + c] = (bf16_t)acc;
}

// ---------------------------------------------------------------------------
// GRU gates + in-place h update + score accumulation
// ---------------------------------------------------------------------------
__global__ void k_gates(const float* __restrict__ gi, const float* __restrict__ gh,
                        float* __restrict__ h, float* __restrict__ score,
                        const float* __restrict__ sw, const float* __restrict__ sb)
{
    int row = blockIdx.x;
    int c = threadIdx.x;   // 64 threads, 48 active
    __shared__ float red[64];
    float contrib = 0.0f;
    if (c < 48) {
        const long base = (long)row * 144;
        float r = gi[base + c] + gh[base + c];
        r = 1.0f / (1.0f + expf(-r));
        float z = gi[base + 48 + c] + gh[base + 48 + c];
        z = 1.0f / (1.0f + expf(-z));
        float nn = tanhf(gi[base + 96 + c] + r * gh[base + 96 + c]);
        float hn = (1.0f - z) * nn + z * h[(long)row * 48 + c];
        h[(long)row * 48 + c] = hn;
        contrib = hn * sw[c];
    }
    red[c] = contrib;
    __syncthreads();
    for (int s = 32; s > 0; s >>= 1) {
        if (c < s) red[c] += red[c + s];
        __syncthreads();
    }
    if (c == 0) score[row] += red[0] + sb[0];
}

// ---------------------------------------------------------------------------
// Output reshapes
// ---------------------------------------------------------------------------
__global__ void k_out_dy(const float* __restrict__ dy, float* __restrict__ out)
{
    int i = blockIdx.x * blockDim.x + threadIdx.x;
    if (i >= K_ * T_ * BN_ * 2) return;
    int c2 = i & 1;
    int bn = (i >> 1) & (BN_ - 1);
    int rem = i >> 10;          // k*40 + t
    int t = rem % T_, kk = rem / T_;
    out[i] = dy[((long)kk * BN_ + bn) * 80 + c2 * 40 + t];
}

__global__ void k_out_score(const float* __restrict__ score, float* __restrict__ out)
{
    int i = blockIdx.x * blockDim.x + threadIdx.x;
    if (i < KBN_) out[i] = score[i];
}

// ---------------------------------------------------------------------------
extern "C" void kernel_launch(void* const* d_in, const int* in_sizes, int n_in,
                              void* d_out, int out_size, void* d_ws, size_t ws_size,
                              hipStream_t stream)
{
    const float* hx      = (const float*)d_in[0];
    const float* cur     = (const float*)d_in[1];
    const float* y_path  = (const float*)d_in[2];
    const float* img     = (const float*)d_in[3];
    const float* c1w     = (const float*)d_in[4];
    const float* c1b     = (const float*)d_in[5];
    const float* c2w     = (const float*)d_in[6];
    const float* c2b     = (const float*)d_in[7];
    const float* fvw     = (const float*)d_in[8];
    const float* fvb     = (const float*)d_in[9];
    const float* scfw    = (const float*)d_in[10];
    const float* scfb    = (const float*)d_in[11];
    const float* wih     = (const float*)d_in[12];
    const float* whh     = (const float*)d_in[13];
    const float* bih     = (const float*)d_in[14];
    const float* bhh     = (const float*)d_in[15];
    const float* dyw     = (const float*)d_in[16];
    const float* dyb     = (const float*)d_in[17];
    const float* sw      = (const float*)d_in[18];
    const float* sb      = (const float*)d_in[19];
    float* out = (float*)d_out;

    // bump allocator over d_ws
    char* p = (char*)d_ws;
    auto grab = [&](size_t bytes) -> void* {
        void* r = (void*)p;
        p += (bytes + 255) & ~(size_t)255;
        return r;
    };
    float*  conv1o = (float*)grab((size_t)B_ * 16 * 80 * 80 * 4);
    float*  fm     = (float*)grab((size_t)B_ * 80 * 80 * 32 * 4);
    float*  h      = (float*)grab((size_t)KBN_ * 48 * 4);
    bf16_t* hb     = (bf16_t*)grab((size_t)KBN_ * 64 * 2);
    bf16_t* Wscf   = (bf16_t*)grab((size_t)36 * 48 * 64 * 2);
    bf16_t* Wih    = (bf16_t*)grab((size_t)144 * 96 * 2);
    bf16_t* Whh    = (bf16_t*)grab((size_t)144 * 64 * 2);
    bf16_t* Wdy    = (bf16_t*)grab((size_t)80 * 64 * 2);
    float*  G      = (float*)grab((size_t)36 * 512 * 48 * 4);
    float*  Aw     = (float*)grab((size_t)KBN_ * 32 * 4);
    int*    Bi     = (int*)grab((size_t)KBN_ * 32 * 4);
    bf16_t* xb     = (bf16_t*)grab((size_t)KBN_ * 96 * 2);
    float*  gi     = (float*)grab((size_t)KBN_ * 144 * 4);
    float*  gh     = (float*)grab((size_t)KBN_ * 144 * 4);
    float*  score  = (float*)grab((size_t)KBN_ * 4);
    float*  dy     = (float*)grab((size_t)KBN_ * 80 * 4);

    // ---- setup ----
    {
        int n = 36 * 48 * 64 + 144 * 96 + 144 * 64 + 80 * 64;
        k_prep_w<<<(n + 255) / 256, 256, 0, stream>>>(scfw, wih, whh, dyw,
                                                      Wscf, Wih, Whh, Wdy);
    }
    k_conv1<<<(B_ * 16 * 80 * 80 + 255) / 256, 256, 0, stream>>>(img, c1w, c1b, conv1o);
    k_conv2<<<(B_ * 32 * 80 * 80 + 255) / 256, 256, 0, stream>>>(conv1o, c2w, c2b, fm);
    k_init<<<(KBN_ * 48 + 255) / 256, 256, 0, stream>>>(hx, h, score);

    // ---- GRU scan over T ----
    for (int t = 0; t < T_; ++t) {
        k_bins<<<(KBN_ + 255) / 256, 256, 0, stream>>>(y_path, t, Aw, Bi);
        k_h2bf16<<<(KBN_ * 64 + 255) / 256, 256, 0, stream>>>(h, hb);
        // G[o] = Hn(512x48,bf16 padK=64) @ Wscf[o]^T     -> (36,512,48)
        wmma_gemm_bf16<<<dim3(512 / 16, 48 / 16, 36), 32, 0, stream>>>(
            hb, 64, Wscf, 64, (long)48 * 64, nullptr, G, 48, (long)512 * 48, 64);
        k_lhalf<<<(KBN_ + 255) / 256, 256, 0, stream>>>(y_path, cur, fm, fvw, fvb, xb, t);
        k_rhalf<<<KBN_, 64, 0, stream>>>(Aw, Bi, G, scfb, xb);
        // gi = x(8192x96) @ Wih^T(144x96) + b_ih
        wmma_gemm_bf16<<<dim3(KBN_ / 16, 144 / 16, 1), 32, 0, stream>>>(
            xb, 96, Wih, 96, 0, bih, gi, 144, 0, 96);
        // gh = h(8192x48 padK=64) @ Whh^T + b_hh
        wmma_gemm_bf16<<<dim3(KBN_ / 16, 144 / 16, 1), 32, 0, stream>>>(
            hb, 64, Whh, 64, 0, bhh, gh, 144, 0, 64);
        k_gates<<<KBN_, 64, 0, stream>>>(gi, gh, h, score, sw, sb);
    }

    // ---- heads ----
    k_h2bf16<<<(KBN_ * 64 + 255) / 256, 256, 0, stream>>>(h, hb);
    wmma_gemm_bf16<<<dim3(KBN_ / 16, 80 / 16, 1), 32, 0, stream>>>(
        hb, 64, Wdy, 64, 0, dyb, dy, 80, 0, 64);
    k_out_dy<<<(K_ * T_ * BN_ * 2 + 255) / 256, 256, 0, stream>>>(dy, out);
    k_out_score<<<(KBN_ + 255) / 256, 256, 0, stream>>>(score, out + (size_t)K_ * T_ * BN_ * 2);
}
